// GatedDeltaPhasorBlock_37082747633875
// MI455X (gfx1250) — matrix operand order repaired and checked by reading, not compiled, not run on hardware
//
#include <hip/hip_runtime.h>
#include <hip/hip_bf16.h>
#include <math.h>

// ---------------- types ----------------
typedef __attribute__((ext_vector_type(16))) __bf16 v16bf;
typedef __attribute__((ext_vector_type(8)))  float  v8f;

__device__ __forceinline__ float bf16u_to_f32(unsigned short u) {
    unsigned int v = ((unsigned int)u) << 16;
    return __builtin_bit_cast(float, v);
}
__device__ __forceinline__ unsigned short f32_to_bf16u(float f) {
    unsigned int u = __builtin_bit_cast(unsigned int, f);
    u += 0x7FFFu + ((u >> 16) & 1u);          // round-to-nearest-even
    return (unsigned short)(u >> 16);
}
__device__ __forceinline__ __bf16 us_as_bf16(unsigned short u) {
    return __builtin_bit_cast(__bf16, u);
}

// -------- CDNA5 async global->LDS helpers (ASYNCcnt-tracked DMA path) --------
__device__ __forceinline__ unsigned lds_off(const void* p) {
    // generic LDS pointer: low 32 bits = wave-relative LDS byte offset
    return (unsigned)(unsigned long long)(size_t)p;
}
__device__ __forceinline__ void async_copy_b128(unsigned lds_byte, const void* g) {
    asm volatile("global_load_async_to_lds_b128 %0, %1, off"
                 :: "v"(lds_byte), "v"((unsigned long long)(size_t)g) : "memory");
}
__device__ __forceinline__ void async_copy_b32(unsigned lds_byte, const void* g) {
    asm volatile("global_load_async_to_lds_b32 %0, %1, off"
                 :: "v"(lds_byte), "v"((unsigned long long)(size_t)g) : "memory");
}
__device__ __forceinline__ void wait_async0() {
    asm volatile("s_wait_asynccnt 0x0" ::: "memory");
}

// ---------------- f32 -> bf16 conversion ----------------
__global__ void cvt_f32_bf16(const float* __restrict__ in,
                             unsigned short* __restrict__ out, long long n) {
    long long i = (long long)blockIdx.x * blockDim.x + threadIdx.x;
    if (i < n) out[i] = f32_to_bf16u(in[i]);
}

// ---------------- WMMA bf16 GEMM, async-LDS double-buffered ----------------
// ACT: 0 = none (f32 out), 1 = gelu (bf16 out), 2 = phase: tanh(.)*pi -> cos,sin (two f32 outs),
//      3 = bias + residual add (f32 out)
// Block = 256 threads = 8 waves (4M x 2N); block tile 128(M) x 64(N);
// each wave owns a 32x32 sub-tile = 2x2 WMMA accumulators.
template <int ACT>
__global__ void __launch_bounds__(256)
gemm_wmma(const unsigned short* __restrict__ A,   // [M,Kd] bf16
          const unsigned short* __restrict__ W,   // [Kd,N] bf16
          const float* __restrict__ bias,         // [N]
          unsigned short* __restrict__ outb,      // bf16 out (ACT 1)
          float* __restrict__ outf,               // f32 out (ACT 0/3) / cos (ACT 2)
          float* __restrict__ outf2,              // sin (ACT 2)
          const float* __restrict__ resid,        // residual (ACT 3)
          int M, int Kd, int N) {
    __shared__ unsigned short As[2][128 * 32];    // 8KB per buffer
    __shared__ unsigned short Bs[2][32 * 64];     // 4KB per buffer

    const int tid  = threadIdx.x;
    const int lane = tid & 31;
    const int wv   = tid >> 5;
    const int wm   = wv & 3;    // wave row (M)
    const int wn   = wv >> 2;   // wave col (N)
    const int mblock = blockIdx.y * 128;
    const int nblock = blockIdx.x * 64;
    const int mrow = lane & 15;
    const int kh   = lane >> 4;

    // stage one 128x32 A tile + 32x64 B tile into LDS buffer nb via async DMA
    auto stage = [&](int nb, int kb) {
        unsigned abase = lds_off(&As[nb][0]);
        unsigned bbase = lds_off(&Bs[nb][0]);
        #pragma unroll
        for (int c = 0; c < 2; ++c) {             // A: 512 chunks of 16B
            int q = tid + c * 256;
            int row = q >> 2, part = q & 3;
            async_copy_b128(abase + row * 64 + part * 16,
                            A + (long long)(mblock + row) * Kd + kb + part * 8);
        }
        {                                         // B: 256 chunks of 16B
            int row = tid >> 3, part = tid & 7;
            async_copy_b128(bbase + row * 128 + part * 16,
                            W + (long long)(kb + row) * N + nblock + part * 8);
        }
    };

    v8f acc[2][2] = {};
    const int nk = Kd >> 5;
    stage(0, 0);
    for (int i = 0; i < nk; ++i) {
        wait_async0();            // this wave's DMA into buf[i&1] landed
        __syncthreads();          // everyone's landed; buf[(i+1)&1] free (prog. order)
        if (i + 1 < nk) stage((i + 1) & 1, (i + 1) * 32);

        const unsigned short* As_c = As[i & 1];
        const unsigned short* Bs_c = Bs[i & 1];
        v16bf av[2], bv[2];
        #pragma unroll
        for (int ti = 0; ti < 2; ++ti) {
            // 16-bit A 16x32 layout: lanes 0-15 K 0-7 & 16-23; lanes 16-31 K 8-15 & 24-31
            const unsigned short* arow = As_c + (wm * 32 + ti * 16 + mrow) * 32;
            #pragma unroll
            for (int v = 0; v < 4; ++v) {
                av[ti][2 * v]     = us_as_bf16(arow[kh * 8 + 2 * v]);
                av[ti][2 * v + 1] = us_as_bf16(arow[kh * 8 + 2 * v + 1]);
                av[ti][2 * v + 8] = us_as_bf16(arow[16 + kh * 8 + 2 * v]);
                av[ti][2 * v + 9] = us_as_bf16(arow[16 + kh * 8 + 2 * v + 1]);
            }
        }
        #pragma unroll
        for (int tj = 0; tj < 2; ++tj) {
            // B 32x16: lane&15 = N col; lanes 0-15 K 0-7 & 16-23; lanes 16-31 K 8-15 & 24-31
            const unsigned short* brow = Bs_c + wn * 32 + tj * 16 + (lane & 15);
            #pragma unroll
            for (int e = 0; e < 16; ++e) bv[tj][e] = us_as_bf16(brow[(((e & 8) << 1) + kh * 8 + (e & 7)) * 64]);
        }
        #pragma unroll
        for (int ti = 0; ti < 2; ++ti)
            #pragma unroll
            for (int tj = 0; tj < 2; ++tj)
                acc[ti][tj] = __builtin_amdgcn_wmma_f32_16x16x32_bf16(
                    false, av[ti], false, bv[tj], (short)0, acc[ti][tj], false, false);
    }

    // C/D layout: VGPR r -> M = r (lanes 0-15) / r+8 (lanes 16-31), N = lane&15
    #pragma unroll
    for (int ti = 0; ti < 2; ++ti) {
        #pragma unroll
        for (int tj = 0; tj < 2; ++tj) {
            const int n = nblock + wn * 32 + tj * 16 + (lane & 15);
            const float bn = bias[n];
            #pragma unroll
            for (int r = 0; r < 8; ++r) {
                int m = mblock + wm * 32 + ti * 16 + r + 8 * kh;
                long long idx = (long long)m * N + n;
                float t = acc[ti][tj][r] + bn;
                if (ACT == 0) {
                    outf[idx] = t;
                } else if (ACT == 1) {
                    float g = 0.5f * t * (1.0f + erff(t * 0.70710678118654752f));
                    outb[idx] = f32_to_bf16u(g);
                } else if (ACT == 2) {
                    float ph = tanhf(t) * 3.14159265358979323846f;
                    outf[idx]  = cosf(ph);
                    outf2[idx] = sinf(ph);
                } else {
                    outf[idx] = t + resid[idx];
                }
            }
        }
    }
}

// ---------------- per-row dot + sigmoid (N=1 projections) ----------------
__global__ void rowdot_sigmoid(const unsigned short* __restrict__ H, // [M,Kd] bf16
                               const float* __restrict__ w,          // [Kd]
                               const float* __restrict__ bias,       // [1]
                               float* __restrict__ out, int M, int Kd) {
    int r = blockIdx.x * blockDim.x + threadIdx.x;
    if (r >= M) return;
    const unsigned short* h = H + (long long)r * Kd;
    float acc = bias[0];
    for (int k = 0; k < Kd; ++k) acc += bf16u_to_f32(h[k]) * w[k];
    out[r] = 1.0f / (1.0f + expf(-acc));
}

// ---------------- sequential phasor scan ----------------
// grid = (D/64, B), block = 64 (2 waves). Each thread owns column d;
// K=64 complex state (128 f32) lives in VGPRs for the whole scan.
// Next step's cos/sin coefficients are async-DMA'd into the alternate
// LDS buffer while the current step's 384-FMA update runs.
__global__ void __launch_bounds__(64)
scan_kernel(const float* __restrict__ cq, const float* __restrict__ sq,
            const float* __restrict__ ck, const float* __restrict__ sk, // [B*L,64]
            const float* __restrict__ alpha, const float* __restrict__ beta, // [B*L]
            const float* __restrict__ value,  // [B*L,D]
            float* __restrict__ outr,         // [B*L,D]
            int L, int D) {
    __shared__ float trig[2][4][64];
    const int tid = threadIdx.x;
    const int b   = blockIdx.y;
    const int d   = blockIdx.x * 64 + tid;
    const long long base = (long long)b * L;

    auto issue = [&](int nb, long long row) {
        long long rk = row * 64 + tid;
        async_copy_b32(lds_off(&trig[nb][0][tid]), cq + rk);
        async_copy_b32(lds_off(&trig[nb][1][tid]), sq + rk);
        async_copy_b32(lds_off(&trig[nb][2][tid]), ck + rk);
        async_copy_b32(lds_off(&trig[nb][3][tid]), sk + rk);
    };

    float mr[64], mi[64];
    #pragma unroll
    for (int k = 0; k < 64; ++k) { mr[k] = 0.0f; mi[k] = 0.0f; }

    const float invSqrtK = 0.125f; // 1/sqrt(64)
    issue(0, base);
    for (int t = 0; t < L; ++t) {
        wait_async0();
        __syncthreads();
        if (t + 1 < L) issue((t + 1) & 1, base + t + 1);

        const float* lcq = trig[t & 1][0];
        const float* lsq = trig[t & 1][1];
        const float* lck = trig[t & 1][2];
        const float* lsk = trig[t & 1][3];

        long long row = base + t;
        float a  = alpha[row];
        float bb = beta[row];
        float v  = value[row * D + d];
        if (t + 1 < L) __builtin_prefetch(&value[(row + 1) * D + d], 0, 0);

        float ret = 0.0f;
        #pragma unroll
        for (int k = 0; k < 64; ++k) ret += mr[k] * lcq[k] + mi[k] * lsq[k];
        ret *= invSqrtK;

        float bc = bb * (v - ret);
        #pragma unroll
        for (int k = 0; k < 64; ++k) {
            mr[k] = a * mr[k] + bc * lck[k];
            mi[k] = a * mi[k] + bc * lsk[k];
        }
        outr[row * D + d] = ret;
    }
}

// ---------------- fused 1/sqrt(t+1) + LayerNorm -> bf16 ----------------
// block = 256 = 8 waves, one wave per row.
__global__ void __launch_bounds__(256)
ln_kernel(const float* __restrict__ rets, const float* __restrict__ g,
          const float* __restrict__ bt, unsigned short* __restrict__ ob,
          int L, int D) {
    const int lane = threadIdx.x & 31;
    const int wv   = threadIdx.x >> 5;
    const long long row = (long long)blockIdx.x * 8 + wv;
    const int t = (int)(row % L);
    const float rn = rsqrtf((float)(t + 1));
    const float* rp = rets + row * D;

    float s = 0.0f, s2 = 0.0f;
    for (int j = lane; j < D; j += 32) {
        float v = rp[j] * rn;
        s += v; s2 += v * v;
    }
    #pragma unroll
    for (int off = 16; off > 0; off >>= 1) {
        s  += __shfl_xor(s, off, 32);
        s2 += __shfl_xor(s2, off, 32);
    }
    float mu  = s / D;
    float var = s2 / D - mu * mu;
    float rstd = rsqrtf(var + 1e-5f);
    unsigned short* op = ob + row * D;
    for (int j = lane; j < D; j += 32) {
        float v = rp[j] * rn;
        op[j] = f32_to_bf16u((v - mu) * rstd * g[j] + bt[j]);
    }
}

// ---------------- host ----------------
extern "C" void kernel_launch(void* const* d_in, const int* in_sizes, int n_in,
                              void* d_out, int out_size, void* d_ws, size_t ws_size,
                              hipStream_t stream) {
    (void)in_sizes; (void)n_in; (void)out_size; (void)ws_size;
    const int Bb = 8, Ll = 4096, Dd = 768, D2 = 384;
    const int M = Bb * Ll; // 32768

    const float* x     = (const float*)d_in[0];
    const float* ke_w1 = (const float*)d_in[1];
    const float* ke_b1 = (const float*)d_in[2];
    const float* ke_w2 = (const float*)d_in[3];
    const float* ke_b2 = (const float*)d_in[4];
    const float* qe_w1 = (const float*)d_in[5];
    const float* qe_b1 = (const float*)d_in[6];
    const float* qe_w2 = (const float*)d_in[7];
    const float* qe_b2 = (const float*)d_in[8];
    const float* v_w   = (const float*)d_in[9];
    const float* v_b   = (const float*)d_in[10];
    const float* a_w1  = (const float*)d_in[11];
    const float* a_b1  = (const float*)d_in[12];
    const float* a_w2  = (const float*)d_in[13];
    const float* a_b2  = (const float*)d_in[14];
    const float* b_w1  = (const float*)d_in[15];
    const float* b_b1  = (const float*)d_in[16];
    const float* b_w2  = (const float*)d_in[17];
    const float* b_b2  = (const float*)d_in[18];
    const float* ln_g  = (const float*)d_in[19];
    const float* ln_b  = (const float*)d_in[20];
    const float* out_w = (const float*)d_in[21];
    const float* out_b = (const float*)d_in[22];
    float* outp = (float*)d_out;

    // workspace carve-up
    char*  ws  = (char*)d_ws;
    size_t off = 0;
    auto carve = [&](size_t bytes) -> char* {
        char* p = ws + off;
        off += (bytes + 255) & ~(size_t)255;
        return p;
    };
    unsigned short* xb   = (unsigned short*)carve((size_t)M * Dd * 2);
    unsigned short* hbuf = (unsigned short*)carve((size_t)M * Dd * 2);
    unsigned short* wke1 = (unsigned short*)carve((size_t)Dd * Dd * 2);
    unsigned short* wqe1 = (unsigned short*)carve((size_t)Dd * Dd * 2);
    unsigned short* wvv  = (unsigned short*)carve((size_t)Dd * Dd * 2);
    unsigned short* wout = (unsigned short*)carve((size_t)Dd * Dd * 2);
    unsigned short* wke2 = (unsigned short*)carve((size_t)Dd * 64 * 2);
    unsigned short* wqe2 = (unsigned short*)carve((size_t)Dd * 64 * 2);
    unsigned short* wa1  = (unsigned short*)carve((size_t)Dd * D2 * 2);
    unsigned short* wb1  = (unsigned short*)carve((size_t)Dd * D2 * 2);
    float* ckb  = (float*)carve((size_t)M * 64 * 4);
    float* skb  = (float*)carve((size_t)M * 64 * 4);
    float* cqb  = (float*)carve((size_t)M * 64 * 4);
    float* sqb  = (float*)carve((size_t)M * 64 * 4);
    float* alph = (float*)carve((size_t)M * 4);
    float* betp = (float*)carve((size_t)M * 4);
    float* valb = (float*)carve((size_t)M * Dd * 4);
    float* rets = (float*)carve((size_t)M * Dd * 4);
    unsigned short* obuf = (unsigned short*)carve((size_t)M * Dd * 2);

    auto cvt = [&](const float* src, unsigned short* dst, long long n) {
        int blocks = (int)((n + 255) / 256);
        cvt_f32_bf16<<<blocks, 256, 0, stream>>>(src, dst, n);
    };
    cvt(x,     xb,   (long long)M * Dd);
    cvt(ke_w1, wke1, (long long)Dd * Dd);
    cvt(qe_w1, wqe1, (long long)Dd * Dd);
    cvt(v_w,   wvv,  (long long)Dd * Dd);
    cvt(out_w, wout, (long long)Dd * Dd);
    cvt(ke_w2, wke2, (long long)Dd * 64);
    cvt(qe_w2, wqe2, (long long)Dd * 64);
    cvt(a_w1,  wa1,  (long long)Dd * D2);
    cvt(b_w1,  wb1,  (long long)Dd * D2);

    dim3 blk(256);
    dim3 gD(Dd / 64, M / 128);   // N=768
    dim3 gK(64 / 64, M / 128);   // N=64
    dim3 gH(D2 / 64, M / 128);   // N=384

    // key path: h = gelu(x@ke_w1+b); phases -> cos/sin
    gemm_wmma<1><<<gD, blk, 0, stream>>>(xb, wke1, ke_b1, hbuf, nullptr, nullptr, nullptr, M, Dd, Dd);
    gemm_wmma<2><<<gK, blk, 0, stream>>>(hbuf, wke2, ke_b2, nullptr, ckb, skb, nullptr, M, Dd, 64);
    // query path
    gemm_wmma<1><<<gD, blk, 0, stream>>>(xb, wqe1, qe_b1, hbuf, nullptr, nullptr, nullptr, M, Dd, Dd);
    gemm_wmma<2><<<gK, blk, 0, stream>>>(hbuf, wqe2, qe_b2, nullptr, cqb, sqb, nullptr, M, Dd, 64);
    // value
    gemm_wmma<0><<<gD, blk, 0, stream>>>(xb, wvv, v_b, nullptr, valb, nullptr, nullptr, M, Dd, Dd);
    // alpha gate
    gemm_wmma<1><<<gH, blk, 0, stream>>>(xb, wa1, a_b1, hbuf, nullptr, nullptr, nullptr, M, Dd, D2);
    rowdot_sigmoid<<<(M + 255) / 256, 256, 0, stream>>>(hbuf, a_w2, a_b2, alph, M, D2);
    // beta gate
    gemm_wmma<1><<<gH, blk, 0, stream>>>(xb, wb1, b_b1, hbuf, nullptr, nullptr, nullptr, M, Dd, D2);
    rowdot_sigmoid<<<(M + 255) / 256, 256, 0, stream>>>(hbuf, b_w2, b_b2, betp, M, D2);

    // sequential scan: 96 blocks x 64 threads, K=64 complex state in VGPRs
    scan_kernel<<<dim3(Dd / 64, Bb), dim3(64), 0, stream>>>(
        cqb, sqb, ckb, skb, alph, betp, valb, rets, Ll, Dd);

    // 1/sqrt(t+1) + layernorm -> bf16
    ln_kernel<<<M / 8, 256, 0, stream>>>(rets, ln_g, ln_b, obuf, Ll, Dd);

    // out projection + bias + residual -> d_out (fp32)
    gemm_wmma<3><<<gD, blk, 0, stream>>>(obuf, wout, out_b, nullptr, outp, nullptr, x, M, Dd, Dd);
}
